// LocationEmbed_75977971466788
// MI455X (gfx1250) — matrix operand order repaired and checked by reading, not compiled, not run on hardware
//
#include <hip/hip_runtime.h>
#include <hip/hip_bf16.h>
#include <math.h>

// ---------------------------------------------------------------------------
// GAT-style GNN forward for gfx1250 (CDNA5, wave32).
// Hot spot: edge MLP GEMM 600k x (258->128) ~ 80 GFLOP -> bf16 WMMA, fused
// with gather + LayerNorm + relu + attention-logit so the 307MB edge tensors
// are never materialized. xl (25.6MB) is L2-resident, so gather traffic stays
// on-chip and WMMA throughput is the ceiling.
//
// Tiling: edge blocks cover 64 edges x 128 cols; each of 8 waves owns a
// 16-col strip and 4 M-tiles, so every B fragment load feeds 4 WMMAs.
// All launch shapes are exact tile multiples (50000=16*3125, 600000=64*9375,
// 64 rows for the head), so inner stores are unguarded straight-line code.
// ---------------------------------------------------------------------------

typedef __bf16 bf16_t;
typedef __attribute__((ext_vector_type(16))) __bf16 v16bf;
typedef __attribute__((ext_vector_type(8)))  __bf16 v8bf;
typedef __attribute__((ext_vector_type(8)))  float  v8f;

#define NN 50000
#define NE 600000
#define F  128
#define KE 288          // 2*128+2 padded up to multiple of 32
#define NG 64
#define EB 64           // edges per block in the edge kernel
#define LN_EPS 1e-5f

__device__ __forceinline__ v8f wmma_bf16(v16bf a, v16bf b, v8f c) {
  // D(16x16,f32) = A(16x32,bf16) * B(32x16,bf16) + C
  return __builtin_amdgcn_wmma_f32_16x16x32_bf16(
      /*neg_a=*/false, a, /*neg_b=*/false, b,
      /*c_mod=*/(short)0, c, /*reuse_a=*/false, /*reuse_b=*/false);
}

__device__ __forceinline__ v16bf join16(v8bf lo, v8bf hi) {
  return __builtin_shufflevector(lo, hi, 0,1,2,3,4,5,6,7,8,9,10,11,12,13,14,15);
}

__device__ __forceinline__ float atomicMaxF(float* addr, float val) {
  unsigned int* ua = (unsigned int*)addr;
  unsigned int old = *ua, assumed;
  do {
    assumed = old;
    if (__uint_as_float(assumed) >= val) break;
    old = atomicCAS(ua, assumed, __float_as_uint(val));
  } while (old != assumed);
  return __uint_as_float(old);
}

// -------------------------------- utility ----------------------------------

__global__ void k_fill(float* __restrict__ p, float v, int n) {
  int i = blockIdx.x * 256 + threadIdx.x;
  if (i < n) p[i] = v;
}

// wt[n*Kpad + k] = w[k*N + n]  (transpose + pad + f32->bf16)
__global__ void k_convert_wt(const float* __restrict__ w, bf16_t* __restrict__ wt,
                             int K, int N, int Kpad) {
  int idx = blockIdx.x * 256 + threadIdx.x;
  if (idx >= N * Kpad) return;
  int nn = idx / Kpad, k = idx - nn * Kpad;
  wt[idx] = (bf16_t)((k < K) ? w[k * N + nn] : 0.0f);
}

// --------------------------- WMMA node linear ------------------------------
// C[nrows x 128] = A[nrows x 128] @ W + bias, W transposed bf16 [128][128].
// Block = 256 threads = 8 waves covering 16 rows x 128 cols; wave -> 16x16.
// Requires nrows % 16 == 0 (true for all call sites).
__global__ void k_node_linear(const float* __restrict__ A,
                              const bf16_t* __restrict__ Bt,
                              const float* __restrict__ bias,
                              float* __restrict__ C, int nrows) {
  __shared__ bf16_t sA[16][136];   // +8 pad against bank conflicts
  int row0 = blockIdx.x * 16;
  int tid  = threadIdx.x;
  // vectorized stage: 16x32 float4 loads -> packed bf16 LDS stores
  for (int idx = tid; idx < 16 * 32; idx += 256) {
    int r = idx >> 5, q = idx & 31;
    float4 v = *(const float4*)(A + (size_t)(row0 + r) * F + 4 * q);
    sA[r][4 * q + 0] = (bf16_t)v.x;
    sA[r][4 * q + 1] = (bf16_t)v.y;
    sA[r][4 * q + 2] = (bf16_t)v.z;
    sA[r][4 * q + 3] = (bf16_t)v.w;
  }
  __syncthreads();

  int lane = tid & 31;
  int n0   = (tid >> 5) * 16;      // wave's N strip
  int m    = lane & 15;
  int hi   = lane >> 4;            // half-wave selector
  v8f acc = {};
  for (int k0 = 0; k0 < F; k0 += 32) {
    v8bf alo = *(const v8bf*)&sA[m][k0 + hi * 8];
    v8bf ahi = *(const v8bf*)&sA[m][k0 + 16 + hi * 8];
    v16bf b = join16(*(const v8bf*)&Bt[(n0 + m) * F + k0 + hi * 8],
                     *(const v8bf*)&Bt[(n0 + m) * F + k0 + 16 + hi * 8]);
    acc = wmma_bf16(join16(alo, ahi), b, acc);
  }
  int n = n0 + m;
  float bn = bias[n];
  #pragma unroll
  for (int r = 0; r < 8; ++r) {
    C[(size_t)(row0 + r + hi * 8) * F + n] = acc[r] + bn;
  }
}

// ----------------------- fused WMMA edge MLP kernel ------------------------
// Per block: 64 edges. Gather [sf|df|aux] (K=258 pad 288) into LDS bf16,
// h_pre = ef @ aw1 + ab1 via WMMA (4 M-tiles per wave reuse each B frag),
// then LN -> relu -> dot(aw2) -> logit, and segment-max atomic for softmax.
// LDS union: 36KB bf16 ef tile, reused as 32KB f32 pre-LN tile (barriered).
__global__ void k_edge_mlp(const float* __restrict__ xl,
                           const int* __restrict__ ei,       // [2*NE]
                           const float* __restrict__ aux,    // [NE*2]
                           const bf16_t* __restrict__ aw1t,  // [128][288] bf16
                           const float* __restrict__ ab1,
                           const float* __restrict__ lng,
                           const float* __restrict__ lnb,
                           const float* __restrict__ aw2,    // [128] f32
                           const float* __restrict__ ab2,    // [1]
                           float* __restrict__ logits,
                           float* __restrict__ segmax) {
  __shared__ __align__(16) char smem[EB * KE * sizeof(bf16_t)];  // 36 KB
  bf16_t (*sE)[KE] = (bf16_t (*)[KE])smem;
  float  (*sH)[F]  = (float  (*)[F])smem;

  int e0  = blockIdx.x * EB;
  int tid = threadIdx.x;

  // --- stage ef: main 256 cols via float4 (col = 4*q for q in [0,64)) ---
  for (int idx = tid; idx < EB * 64; idx += 256) {
    int r = idx >> 6, q = idx & 63;
    int e = e0 + r;
    int node = (q < 32) ? ei[e] : ei[NE + e];
    int c    = (q < 32) ? (4 * q) : (4 * q - 128);
    float4 v = *(const float4*)(xl + (size_t)node * F + c);
    int col = 4 * q;
    sE[r][col + 0] = (bf16_t)v.x;
    sE[r][col + 1] = (bf16_t)v.y;
    sE[r][col + 2] = (bf16_t)v.z;
    sE[r][col + 3] = (bf16_t)v.w;
  }
  // --- tail cols 256..287: aux (2) + zero pad (30) ---
  for (int idx = tid; idx < EB * 32; idx += 256) {
    int r = idx >> 5, k = 256 + (idx & 31);
    float v = (k < 258) ? aux[(size_t)(e0 + r) * 2 + (k - 256)] : 0.0f;
    sE[r][k] = (bf16_t)v;
  }
  __syncthreads();

  int lane = tid & 31;
  int n0   = (tid >> 5) * 16;
  int m    = lane & 15;
  int hi   = lane >> 4;
  v8f acc0 = {}, acc1 = {}, acc2 = {}, acc3 = {};
  for (int k0 = 0; k0 < KE; k0 += 32) {
    v16bf b = join16(*(const v8bf*)&aw1t[(n0 + m) * KE + k0 + hi * 8],
                     *(const v8bf*)&aw1t[(n0 + m) * KE + k0 + 16 + hi * 8]);
    v8bf a0l = *(const v8bf*)&sE[m     ][k0 + hi * 8];
    v8bf a0h = *(const v8bf*)&sE[m     ][k0 + 16 + hi * 8];
    acc0 = wmma_bf16(join16(a0l, a0h), b, acc0);
    v8bf a1l = *(const v8bf*)&sE[m + 16][k0 + hi * 8];
    v8bf a1h = *(const v8bf*)&sE[m + 16][k0 + 16 + hi * 8];
    acc1 = wmma_bf16(join16(a1l, a1h), b, acc1);
    v8bf a2l = *(const v8bf*)&sE[m + 32][k0 + hi * 8];
    v8bf a2h = *(const v8bf*)&sE[m + 32][k0 + 16 + hi * 8];
    acc2 = wmma_bf16(join16(a2l, a2h), b, acc2);
    v8bf a3l = *(const v8bf*)&sE[m + 48][k0 + hi * 8];
    v8bf a3h = *(const v8bf*)&sE[m + 48][k0 + 16 + hi * 8];
    acc3 = wmma_bf16(join16(a3l, a3h), b, acc3);
  }
  __syncthreads();   // all sE reads done before overwriting as sH

  int n = n0 + m;
  float bn = ab1[n];
  #pragma unroll
  for (int r = 0; r < 8; ++r) {
    sH[r + hi * 8     ][n] = acc0[r] + bn;
    sH[r + hi * 8 + 16][n] = acc1[r] + bn;
    sH[r + hi * 8 + 32][n] = acc2[r] + bn;
    sH[r + hi * 8 + 48][n] = acc3[r] + bn;
  }
  __syncthreads();

  // --- LN + relu + logit: 4 lanes per edge, width-4 butterfly reductions ---
  int el   = tid >> 2;        // edge 0..63
  int part = tid & 3;         // 32-col slice
  int c0   = part * 32;
  float s1 = 0.0f, s2 = 0.0f;
  #pragma unroll
  for (int j = 0; j < 32; ++j) {
    float v = sH[el][c0 + j];
    s1 += v;
    s2 += v * v;
  }
  s1 += __shfl_xor(s1, 1, 4);  s2 += __shfl_xor(s2, 1, 4);
  s1 += __shfl_xor(s1, 2, 4);  s2 += __shfl_xor(s2, 2, 4);
  float mean = s1 * (1.0f / 128.0f);
  float var  = s2 * (1.0f / 128.0f) - mean * mean;
  float rinv = rsqrtf(var + LN_EPS);
  float dot = 0.0f;
  #pragma unroll
  for (int j = 0; j < 32; ++j) {
    int k = c0 + j;
    float h = (sH[el][k] - mean) * rinv * lng[k] + lnb[k];
    h = h > 0.0f ? h : 0.0f;
    dot += h * aw2[k];
  }
  dot += __shfl_xor(dot, 1, 4);
  dot += __shfl_xor(dot, 2, 4);
  if (part == 0) {
    int e = e0 + el;
    float logit = dot + ab2[0];
    logits[e] = logit;
    atomicMaxF(&segmax[ei[NE + e]], logit);
  }
}

// -------------------- segment softmax / scatter-mean -----------------------

__global__ void k_edge_exp(const int* __restrict__ ei,
                           float* __restrict__ logits,
                           const float* __restrict__ segmax,
                           float* __restrict__ segsum,
                           float* __restrict__ cnt) {
  int e = blockIdx.x * 256 + threadIdx.x;
  if (e >= NE) return;
  int d = ei[NE + e];
  float ex = __expf(logits[e] - segmax[d]);
  logits[e] = ex;                 // overwrite with numerator
  atomicAdd(&segsum[d], ex);
  atomicAdd(&cnt[d], 1.0f);
}

__global__ void k_edge_agg(const int* __restrict__ ei,
                           const float* __restrict__ enum_,
                           const float* __restrict__ segsum,
                           const float* __restrict__ xl,
                           float* __restrict__ agg) {
  long long idx = (long long)blockIdx.x * 256 + threadIdx.x;
  if (idx >= (long long)NE * F) return;
  int e = (int)(idx >> 7), c = (int)(idx & 127);
  int d = ei[NE + e], s = ei[e];
  float att = enum_[e] / segsum[d];
  atomicAdd(&agg[(size_t)d * F + c], xl[(size_t)s * F + c] * att);
}

// ------------------------------- finalize ----------------------------------

__global__ void k_finalize_relu(const float* __restrict__ agg,
                                const float* __restrict__ cnt,
                                const float* __restrict__ xl,
                                float* __restrict__ out) {
  int idx = blockIdx.x * 256 + threadIdx.x;
  if (idx >= NN * F) return;
  int n = idx >> 7;
  float v = agg[idx] / fmaxf(cnt[n], 1.0f) + xl[idx];
  out[idx] = fmaxf(v, 0.0f);
}

// One 128-thread block per node: scatter-mean residual + LayerNorm +
// graph-level segment-max pool (fusion) via atomic max.
__global__ void k_finalize_ln_pool(const float* __restrict__ agg,
                                   const float* __restrict__ cnt,
                                   const float* __restrict__ xl,
                                   const float* __restrict__ g,
                                   const float* __restrict__ b,
                                   const int* __restrict__ batch,
                                   float* __restrict__ hout,
                                   float* __restrict__ fusion) {
  __shared__ float red[128];
  int n = blockIdx.x, c = threadIdx.x;
  float v = agg[(size_t)n * F + c] / fmaxf(cnt[n], 1.0f) + xl[(size_t)n * F + c];
  red[c] = v;
  __syncthreads();
  for (int s = 64; s > 0; s >>= 1) { if (c < s) red[c] += red[c + s]; __syncthreads(); }
  float mean = red[0] * (1.0f / 128.0f);
  __syncthreads();
  float d = v - mean;
  red[c] = d * d;
  __syncthreads();
  for (int s = 64; s > 0; s >>= 1) { if (c < s) red[c] += red[c + s]; __syncthreads(); }
  float var = red[0] * (1.0f / 128.0f);
  float hn = (v - mean) * rsqrtf(var + LN_EPS) * g[c] + b[c];
  hout[(size_t)n * F + c] = hn;
  atomicMaxF(&fusion[(size_t)batch[n] * F + c], hn);
}

// -------------------------------- host side --------------------------------

static void run_conv(hipStream_t stream, const float* hin,
                     const int* ei, const float* aux,
                     const bf16_t* wlint, const float* b_lin,
                     const bf16_t* aw1t, const float* ab1,
                     const float* lng, const float* lnb,
                     const float* aw2, const float* ab2,
                     float* xl, float* agg, float* logits,
                     float* segmax, float* segsum, float* cnt) {
  k_fill<<<(NN + 255) / 256, 256, 0, stream>>>(segmax, -INFINITY, NN);
  k_fill<<<(NN + 255) / 256, 256, 0, stream>>>(segsum, 0.0f, NN);
  k_fill<<<(NN + 255) / 256, 256, 0, stream>>>(cnt, 0.0f, NN);
  k_fill<<<(NN * F + 255) / 256, 256, 0, stream>>>(agg, 0.0f, NN * F);
  k_node_linear<<<NN / 16, 256, 0, stream>>>(hin, wlint, b_lin, xl, NN);
  k_edge_mlp<<<NE / EB, 256, 0, stream>>>(xl, ei, aux, aw1t, ab1, lng, lnb,
                                          aw2, ab2, logits, segmax);
  k_edge_exp<<<(NE + 255) / 256, 256, 0, stream>>>(ei, logits, segmax, segsum, cnt);
  k_edge_agg<<<(NE * F) / 256, 256, 0, stream>>>(ei, logits, segsum, xl, agg);
}

extern "C" void kernel_launch(void* const* d_in, const int* in_sizes, int n_in,
                              void* d_out, int out_size, void* d_ws, size_t ws_size,
                              hipStream_t stream) {
  const float* x      = (const float*)d_in[0];
  const int*   ei     = (const int*)  d_in[1];
  const float* aux    = (const float*)d_in[2];
  const int*   batch  = (const int*)  d_in[3];
  const float* w_lin1 = (const float*)d_in[4];
  const float* b_lin1 = (const float*)d_in[5];
  const float* aw1_1  = (const float*)d_in[6];
  const float* ab1_1  = (const float*)d_in[7];
  const float* lng1   = (const float*)d_in[8];
  const float* lnb1   = (const float*)d_in[9];
  const float* aw2_1  = (const float*)d_in[10];
  const float* ab2_1  = (const float*)d_in[11];
  const float* w_lin2 = (const float*)d_in[12];
  const float* b_lin2 = (const float*)d_in[13];
  const float* aw1_2  = (const float*)d_in[14];
  const float* ab1_2  = (const float*)d_in[15];
  const float* lng2   = (const float*)d_in[16];
  const float* lnb2   = (const float*)d_in[17];
  const float* aw2_2  = (const float*)d_in[18];
  const float* ab2_2  = (const float*)d_in[19];
  const float* cls_w  = (const float*)d_in[20];
  const float* cls_b  = (const float*)d_in[21];
  const float* norm_g = (const float*)d_in[22];
  const float* norm_b = (const float*)d_in[23];

  // ---- carve workspace (256B aligned slices) ----
  char* p = (char*)d_ws;
  auto carve = [&](size_t bytes) -> void* {
    void* r = (void*)p;
    p += (bytes + 255) & ~(size_t)255;
    return r;
  };
  float*  xl      = (float*) carve((size_t)NN * F * 4);
  float*  hbuf    = (float*) carve((size_t)NN * F * 4);
  float*  agg     = (float*) carve((size_t)NN * F * 4);
  float*  logits  = (float*) carve((size_t)NE * 4);
  float*  segmax  = (float*) carve((size_t)NN * 4);
  float*  segsum  = (float*) carve((size_t)NN * 4);
  float*  cnt     = (float*) carve((size_t)NN * 4);
  float*  fusion  = (float*) carve((size_t)NG * F * 4);
  bf16_t* wlin1t  = (bf16_t*)carve((size_t)F * F * 2);
  bf16_t* wlin2t  = (bf16_t*)carve((size_t)F * F * 2);
  bf16_t* clswt   = (bf16_t*)carve((size_t)F * F * 2);
  bf16_t* aw11t   = (bf16_t*)carve((size_t)F * KE * 2);
  bf16_t* aw12t   = (bf16_t*)carve((size_t)F * KE * 2);

  // ---- convert weights to transposed bf16 (padded K) ----
  int nsq = F * F;
  int nke = F * KE;
  k_convert_wt<<<(nsq + 255) / 256, 256, 0, stream>>>(w_lin1, wlin1t, 128, 128, 128);
  k_convert_wt<<<(nsq + 255) / 256, 256, 0, stream>>>(w_lin2, wlin2t, 128, 128, 128);
  k_convert_wt<<<(nsq + 255) / 256, 256, 0, stream>>>(cls_w,  clswt,  128, 128, 128);
  k_convert_wt<<<(nke + 255) / 256, 256, 0, stream>>>(aw1_1,  aw11t,  258, 128, KE);
  k_convert_wt<<<(nke + 255) / 256, 256, 0, stream>>>(aw1_2,  aw12t,  258, 128, KE);

  // ---- conv1: x -> hbuf (relu) ----
  run_conv(stream, x, ei, aux, wlin1t, b_lin1, aw11t, ab1_1, lng1, lnb1,
           aw2_1, ab2_1, xl, agg, logits, segmax, segsum, cnt);
  k_finalize_relu<<<(NN * F) / 256, 256, 0, stream>>>(agg, cnt, xl, hbuf);

  // ---- conv2: hbuf -> (LN -> d_out h part) + fusion pool ----
  run_conv(stream, hbuf, ei, aux, wlin2t, b_lin2, aw12t, ab1_2, lng2, lnb2,
           aw2_2, ab2_2, xl, agg, logits, segmax, segsum, cnt);
  k_fill<<<(NG * F + 255) / 256, 256, 0, stream>>>(fusion, -INFINITY, NG * F);
  float* h_out = (float*)d_out;                      // [NN*F]
  k_finalize_ln_pool<<<NN, 128, 0, stream>>>(agg, cnt, xl, norm_g, norm_b, batch,
                                             h_out, fusion);

  // ---- classifier head: cls = fusion @ cls_w + cls_b (WMMA) ----
  float* cls_out = h_out + (size_t)NN * F;           // [NG*F]
  k_node_linear<<<NG / 16, 256, 0, stream>>>(fusion, clswt, cls_b, cls_out, NG);
}
